// CrossAttnUGCA_5497558139216
// MI455X (gfx1250) — matrix operand order repaired and checked by reading, not compiled, not run on hardware
//
#include <hip/hip_runtime.h>
#include <hip/hip_bf16.h>
#include <math.h>

// ---------------- problem constants ----------------
#define DD    768
#define HH    8
#define HD    96
#define BB    32
#define LL    1024
#define MTOT  (BB*LL)          // 32768 token rows
#define K4D   3072             // 4*D gating feature width
#define NG    192              // gating hidden width (D/4)
#define LN_EPS 1e-5f

typedef __bf16 bf16;
typedef __attribute__((ext_vector_type(16))) __bf16 v16bf;
typedef __attribute__((ext_vector_type(8)))  __bf16 v8bf;
typedef __attribute__((ext_vector_type(8)))  float  v8f;

static __device__ __forceinline__ bf16  f2bf(float f) { return (bf16)f; }
static __device__ __forceinline__ float bf2f(bf16 x)  { return (float)x; }

// D = A(16x32 bf16) * B(32x16 bf16) + C(16x16 f32)
static __device__ __forceinline__ v8f wmma_bf16(v16bf a, v16bf b, v8f c) {
  return __builtin_amdgcn_wmma_f32_16x16x32_bf16(
      /*neg_a=*/false, a, /*neg_b=*/false, b,
      /*c_mod=*/(short)0, c, /*reuse_a=*/false, /*reuse_b=*/false);
}

// ---------------- tiny prep kernels ----------------
__global__ void cvt_bf16_kernel(const float* __restrict__ in,
                                bf16* __restrict__ out, int n) {
  int i = blockIdx.x * blockDim.x + threadIdx.x;
  int stride = gridDim.x * blockDim.x;
  for (; i < n; i += stride) out[i] = f2bf(in[i]);
}

// in [K,N] f32 row-major  ->  out [N,K] bf16 row-major (B-matrix friendly)
__global__ void transpose_bf16_kernel(const float* __restrict__ in,
                                      bf16* __restrict__ out, int K, int N) {
  int i = blockIdx.x * blockDim.x + threadIdx.x;
  int stride = gridDim.x * blockDim.x;
  int total = K * N;
  for (; i < total; i += stride) {
    int n = i / K, k = i - n * K;
    out[i] = f2bf(in[k * N + n]);
  }
}

// q = q_vec @ Wq + bq   (tiny: 36 MFLOP)
__global__ void __launch_bounds__(256)
qproj_kernel(const float* __restrict__ q_vec, const float* __restrict__ Wq,
             const float* __restrict__ bq, float* __restrict__ qp) {
  __shared__ float qs[DD];
  const int b = blockIdx.x, tid = threadIdx.x;
  for (int i = tid; i < DD; i += 256) qs[i] = q_vec[b * DD + i];
  __syncthreads();
  for (int n = tid; n < DD; n += 256) {
    float acc = bq[n];
    for (int d = 0; d < DD; ++d) acc = fmaf(qs[d], Wq[d * DD + n], acc);
    qp[b * DD + n] = acc;
  }
}

// ---------------- K/V projection: [32768,768] @ [768,768] x2, bf16 WMMA ----------------
// grid = (MTOT/32, D/128), block = 128 (4 waves).
// Each wave: 32 rows x 32 cols of BOTH K and V -> 8 accumulators; every B
// fragment feeds two WMMAs (rows 0-15 and 16-31), halving weight traffic and
// giving the scheduler 8 independent WMMA chains to hide load latency.
__global__ void __launch_bounds__(128)
kvproj_kernel(const bf16* __restrict__ kvb,
              const bf16* __restrict__ WkT, const bf16* __restrict__ WvT,
              const float* __restrict__ biasK, const float* __restrict__ biasV,
              bf16* __restrict__ Kp, bf16* __restrict__ Vp) {
  const int mt   = blockIdx.x;           // 32-row slab
  const int ng   = blockIdx.y;           // 128-col slab
  const int wave = threadIdx.x >> 5;
  const int lane = threadIdx.x & 31;
  const int nbase = ng * 128 + wave * 32;

  v8f aK00 = {}, aK01 = {}, aK10 = {}, aK11 = {};
  v8f aV00 = {}, aV01 = {}, aV10 = {}, aV11 = {};

  // A as loaded here (contiguous k, not AMD's table order): lanes 0-15 row=lane hold K 0..15;
  // lanes 16-31 row=lane-16 hold K 16..31.
  const int arow0 = mt * 32 + (lane & 15);       // rows 0-15 of slab
  const int asel  = (lane >> 4) * 16;
  const bf16* arowp0 = kvb + (size_t)arow0 * DD;
  const bf16* arowp1 = arowp0 + (size_t)16 * DD; // rows 16-31 of slab
  // B as loaded here (same contiguous k as A): lanes 0-15 col=lane hold K 0..15;
  // lanes 16-31 col=lane-16 hold K 16..31. WT is [N][K] row-major.
  const int bcol0 = nbase + (lane & 15);
  const int bcol1 = bcol0 + 16;
  const int bksel = (lane >> 4) * 16;

  for (int k = 0; k < DD; k += 32) {
    v8bf a0lo = *(const v8bf*)(arowp0 + k + asel);
    v8bf a0hi = *(const v8bf*)(arowp0 + k + asel + 8);
    v8bf a1lo = *(const v8bf*)(arowp1 + k + asel);
    v8bf a1hi = *(const v8bf*)(arowp1 + k + asel + 8);
    v16bf a0 = __builtin_shufflevector(a0lo, a0hi,
        0,1,2,3,4,5,6,7,8,9,10,11,12,13,14,15);
    v16bf a1 = __builtin_shufflevector(a1lo, a1hi,
        0,1,2,3,4,5,6,7,8,9,10,11,12,13,14,15);
    v16bf bK0 = *(const v16bf*)(WkT + (size_t)bcol0 * DD + k + bksel);
    v16bf bK1 = *(const v16bf*)(WkT + (size_t)bcol1 * DD + k + bksel);
    v16bf bV0 = *(const v16bf*)(WvT + (size_t)bcol0 * DD + k + bksel);
    v16bf bV1 = *(const v16bf*)(WvT + (size_t)bcol1 * DD + k + bksel);
    aK00 = wmma_bf16(a0, bK0, aK00);
    aK10 = wmma_bf16(a1, bK0, aK10);
    aK01 = wmma_bf16(a0, bK1, aK01);
    aK11 = wmma_bf16(a1, bK1, aK11);
    aV00 = wmma_bf16(a0, bV0, aV00);
    aV10 = wmma_bf16(a1, bV0, aV10);
    aV01 = wmma_bf16(a0, bV1, aV01);
    aV11 = wmma_bf16(a1, bV1, aV11);
  }

  // C/D layout: lanes 0-15 col=lane rows r; lanes 16-31 col=lane-16 rows r+8.
  const int rbase0 = mt * 32 + ((lane < 16) ? 0 : 8);
  const int rbase1 = rbase0 + 16;
  const int c0 = nbase + (lane & 15), c1 = c0 + 16;
  const float bK0s = biasK[c0], bK1s = biasK[c1];
  const float bV0s = biasV[c0], bV1s = biasV[c1];
#pragma unroll
  for (int r = 0; r < 8; ++r) {
    size_t row0 = (size_t)(rbase0 + r) * DD;
    size_t row1 = (size_t)(rbase1 + r) * DD;
    Kp[row0 + c0] = f2bf(aK00[r] + bK0s);
    Kp[row0 + c1] = f2bf(aK01[r] + bK1s);
    Kp[row1 + c0] = f2bf(aK10[r] + bK0s);
    Kp[row1 + c1] = f2bf(aK11[r] + bK1s);
    Vp[row0 + c0] = f2bf(aV00[r] + bV0s);
    Vp[row0 + c1] = f2bf(aV01[r] + bV1s);
    Vp[row1 + c0] = f2bf(aV10[r] + bV0s);
    Vp[row1 + c1] = f2bf(aV11[r] + bV1s);
  }
}

// ---------------- gating MLP: feat(3072) @ Wg1 -> relu -> @Wg2 -> softplus -> g ----------------
// grid = MTOT/16, block = 128 (4 waves, each owns 3 of the 12 N-tiles).
__global__ void __launch_bounds__(128)
gating_kernel(const float* __restrict__ q_vec, const float* __restrict__ kv,
              const bf16* __restrict__ Wg1T, const float* __restrict__ bg1,
              const float* __restrict__ Wg2, const float* __restrict__ bg2,
              float* __restrict__ log_g, float* __restrict__ g_out) {
  __shared__ float qs[DD];
  __shared__ float kvt[16][DD];
  __shared__ float hbuf[16][NG];

  const int mt  = blockIdx.x;
  const int b   = mt >> 6;            // 64 tiles per batch row (1024/16)
  const int l0  = (mt & 63) * 16;
  const int tid = threadIdx.x;
  const int wave = tid >> 5, lane = tid & 31;

  for (int i = tid; i < DD; i += 128) qs[i] = q_vec[b * DD + i];
  for (int i = tid; i < 16 * DD; i += 128) {
    int r = i / DD, c = i - r * DD;
    kvt[r][c] = kv[(size_t)(b * LL + l0 + r) * DD + c];
  }
  __syncthreads();

  v8f acc0 = {}, acc1 = {}, acc2 = {};
  const int r     = lane & 15;
  const int ksel  = (lane >> 4) * 16;
  const int nt0   = wave * 3;
  const int bcol0 = nt0 * 16 + (lane & 15);
  const int bksel = (lane >> 4) * 16;

  for (int kg = 0; kg < K4D; kg += 32) {
    const int blk = kg / DD;              // which interaction block
    const int cb  = kg - blk * DD;        // feature offset inside block
    float fv[16];
#pragma unroll
    for (int j = 0; j < 16; ++j) {
      int cc = cb + ksel + j; // K{0..15} contiguous
      float q = qs[cc], t = kvt[r][cc];
      float f;
      if      (blk == 0) f = q;
      else if (blk == 1) f = t;
      else if (blk == 2) f = fabsf(q - t);
      else               f = q * t;
      fv[j] = f;
    }
    v16bf a;
#pragma unroll
    for (int j = 0; j < 16; ++j) a[j] = f2bf(fv[j]);
    v16bf b0 = *(const v16bf*)(Wg1T + (size_t)(bcol0)      * K4D + kg + bksel);
    v16bf b1 = *(const v16bf*)(Wg1T + (size_t)(bcol0 + 16) * K4D + kg + bksel);
    v16bf b2 = *(const v16bf*)(Wg1T + (size_t)(bcol0 + 32) * K4D + kg + bksel);
    acc0 = wmma_bf16(a, b0, acc0);
    acc1 = wmma_bf16(a, b1, acc1);
    acc2 = wmma_bf16(a, b2, acc2);
  }

  // epilogue: bias + relu -> LDS h, then per-token 192-dot with Wg2
  const int rowb = (lane < 16) ? 0 : 8;
  const int cA   = lane & 15;
#pragma unroll
  for (int rr = 0; rr < 8; ++rr) {
    int m = rowb + rr, n0 = nt0 * 16 + cA;
    hbuf[m][n0]      = fmaxf(acc0[rr] + bg1[n0],      0.f);
    hbuf[m][n0 + 16] = fmaxf(acc1[rr] + bg1[n0 + 16], 0.f);
    hbuf[m][n0 + 32] = fmaxf(acc2[rr] + bg1[n0 + 32], 0.f);
  }
  __syncthreads();

  if (tid < 16) {
    float e = bg2[0];
    for (int n = 0; n < NG; ++n) e = fmaf(hbuf[tid][n], Wg2[n], e);
    float sp = (e > 20.f) ? e : log1pf(expf(e));        // softplus
    float g  = sp / (sp + 1.f + 1e-8f);
    g = fminf(fmaxf(g, 0.01f), 0.99f);
    int idx = b * LL + l0 + tid;
    g_out[idx] = g;
    log_g[idx] = logf(g);
  }
}

// mean(log_g) per batch (over ALL L positions, matching reference)
__global__ void __launch_bounds__(256)
meanlog_kernel(const float* __restrict__ log_g, float* __restrict__ mlg) {
  __shared__ float red[256];
  const int b = blockIdx.x, tid = threadIdx.x;
  float s = 0.f;
  for (int l = tid; l < LL; l += 256) s += log_g[b * LL + l];
  red[tid] = s; __syncthreads();
  for (int off = 128; off > 0; off >>= 1) {
    if (tid < off) red[tid] += red[tid + off];
    __syncthreads();
  }
  if (tid == 0) mlg[b] = red[0] / (float)LL;
}

// ---------------- single-query attention per (b,h) ----------------
__global__ void __launch_bounds__(128)
attn_kernel(const float* __restrict__ qp, const bf16* __restrict__ Kp,
            const bf16* __restrict__ Vp, const unsigned char* __restrict__ mask,
            const float* __restrict__ log_g, const float* __restrict__ mlg,
            float* __restrict__ ctx) {
  __shared__ float sc[LL];
  __shared__ float qh[HD];
  __shared__ float red[128];
  const int b = blockIdx.x, h = blockIdx.y, tid = threadIdx.x;
  if (tid < HD) qh[tid] = qp[b * DD + h * HD + tid];
  __syncthreads();

  const float scale = 0.102062072615966f;   // 1/sqrt(96)
  const float mg = mlg[b];
  for (int i = 0; i < LL / 128; ++i) {
    int l = tid + i * 128;
    const bf16* kp = Kp + (size_t)(b * LL + l) * DD + h * HD;
    float s = 0.f;
    for (int d = 0; d < HD; ++d) s = fmaf(qh[d], bf2f(kp[d]), s);
    s = s * scale + (log_g[b * LL + l] - mg);           // BETA = 1
    sc[l] = mask[b * LL + l] ? s : -__builtin_inff();
  }
  __syncthreads();

  float mx = -3.4e38f;
  for (int i = 0; i < LL / 128; ++i) mx = fmaxf(mx, sc[tid + i * 128]);
  red[tid] = mx; __syncthreads();
  for (int off = 64; off > 0; off >>= 1) {
    if (tid < off) red[tid] = fmaxf(red[tid], red[tid + off]);
    __syncthreads();
  }
  mx = red[0]; __syncthreads();

  float ss = 0.f;
  for (int i = 0; i < LL / 128; ++i) {
    int l = tid + i * 128;
    float w = expf(sc[l] - mx);   // exp(-inf)=0 for masked
    sc[l] = w; ss += w;
  }
  __syncthreads();
  red[tid] = ss; __syncthreads();
  for (int off = 64; off > 0; off >>= 1) {
    if (tid < off) red[tid] += red[tid + off];
    __syncthreads();
  }
  const float tot = red[0];
  __syncthreads();

  if (tid < HD) {
    float acc = 0.f;
    for (int l = 0; l < LL; ++l)
      acc = fmaf(sc[l], bf2f(Vp[(size_t)(b * LL + l) * DD + h * HD + tid]), acc);
    ctx[b * DD + h * HD + tid] = acc / tot;
  }
}

// ---------------- out projection + residual + LayerNorm ----------------
__global__ void __launch_bounds__(256)
outln_kernel(const float* __restrict__ q_vec, const float* __restrict__ ctx,
             const float* __restrict__ Wo, const float* __restrict__ bo,
             const float* __restrict__ ln_g, const float* __restrict__ ln_b,
             float* __restrict__ out) {
  __shared__ float cs[DD];
  __shared__ float xs[DD];
  __shared__ float red[256];
  const int b = blockIdx.x, tid = threadIdx.x;
  for (int i = tid; i < DD; i += 256) cs[i] = ctx[b * DD + i];
  __syncthreads();
  for (int n = tid; n < DD; n += 256) {
    float y = bo[n];
    for (int d = 0; d < DD; ++d) y = fmaf(cs[d], Wo[d * DD + n], y);
    xs[n] = q_vec[b * DD + n] + y;
  }
  __syncthreads();
  float s = 0.f;
  for (int i = tid; i < DD; i += 256) s += xs[i];
  red[tid] = s; __syncthreads();
  for (int off = 128; off > 0; off >>= 1) {
    if (tid < off) red[tid] += red[tid + off];
    __syncthreads();
  }
  const float mu = red[0] / (float)DD;
  __syncthreads();
  float v = 0.f;
  for (int i = tid; i < DD; i += 256) { float d0 = xs[i] - mu; v += d0 * d0; }
  red[tid] = v; __syncthreads();
  for (int off = 128; off > 0; off >>= 1) {
    if (tid < off) red[tid] += red[tid + off];
    __syncthreads();
  }
  const float rs = rsqrtf(red[0] / (float)DD + LN_EPS);
  __syncthreads();
  for (int n = tid; n < DD; n += 256)
    out[b * DD + n] = (xs[n] - mu) * rs * ln_g[n] + ln_b[n];
}

// ---------------- host launcher ----------------
extern "C" void kernel_launch(void* const* d_in, const int* in_sizes, int n_in,
                              void* d_out, int out_size, void* d_ws, size_t ws_size,
                              hipStream_t stream) {
  const float* q_vec = (const float*)d_in[0];
  const float* kv    = (const float*)d_in[1];
  const unsigned char* mask = (const unsigned char*)d_in[2]; // jnp bool -> 1B
  const float* Wg1 = (const float*)d_in[3];
  const float* bg1 = (const float*)d_in[4];
  const float* Wg2 = (const float*)d_in[5];
  const float* bg2 = (const float*)d_in[6];
  const float* Wq  = (const float*)d_in[7];
  const float* bq  = (const float*)d_in[8];
  const float* Wk  = (const float*)d_in[9];
  const float* bk  = (const float*)d_in[10];
  const float* Wv  = (const float*)d_in[11];
  const float* bv  = (const float*)d_in[12];
  const float* Wo  = (const float*)d_in[13];
  const float* bo  = (const float*)d_in[14];
  const float* lng = (const float*)d_in[15];
  const float* lnb = (const float*)d_in[16];
  float* out = (float*)d_out;          // [B*D fused][B*L g_tok]

  char* ws = (char*)d_ws;
  size_t off = 0;
  auto alloc = [&](size_t bytes) -> void* {
    void* p = ws + off;
    off = (off + bytes + 255) & ~(size_t)255;
    return p;
  };
  bf16*  kvb  = (bf16*) alloc((size_t)MTOT * DD * 2);   // 48 MB
  bf16*  WkT  = (bf16*) alloc((size_t)DD * DD * 2);
  bf16*  WvT  = (bf16*) alloc((size_t)DD * DD * 2);
  bf16*  Wg1T = (bf16*) alloc((size_t)NG * K4D * 2);
  bf16*  Kp   = (bf16*) alloc((size_t)MTOT * DD * 2);   // 48 MB
  bf16*  Vp   = (bf16*) alloc((size_t)MTOT * DD * 2);   // 48 MB
  float* logg = (float*)alloc((size_t)BB * LL * 4);
  float* mlg  = (float*)alloc((size_t)BB * 4);
  float* qp   = (float*)alloc((size_t)BB * DD * 4);
  float* ctx  = (float*)alloc((size_t)BB * DD * 4);
  (void)ws_size; // ~146 MB total assumed available

  cvt_bf16_kernel<<<2048, 256, 0, stream>>>(kv, kvb, MTOT * DD);
  transpose_bf16_kernel<<<1024, 256, 0, stream>>>(Wk, WkT, DD, DD);
  transpose_bf16_kernel<<<1024, 256, 0, stream>>>(Wv, WvT, DD, DD);
  transpose_bf16_kernel<<<1024, 256, 0, stream>>>(Wg1, Wg1T, K4D, NG);
  qproj_kernel<<<BB, 256, 0, stream>>>(q_vec, Wq, bq, qp);
  kvproj_kernel<<<dim3(MTOT / 32, DD / 128), 128, 0, stream>>>(
      kvb, WkT, WvT, bk, bv, Kp, Vp);
  gating_kernel<<<MTOT / 16, 128, 0, stream>>>(
      q_vec, kv, Wg1T, bg1, Wg2, bg2, logg, out + BB * DD);
  meanlog_kernel<<<BB, 256, 0, stream>>>(logg, mlg);
  attn_kernel<<<dim3(BB, HH), 128, 0, stream>>>(qp, Kp, Vp, mask, logg, mlg, ctx);
  outln_kernel<<<BB, 256, 0, stream>>>(q_vec, ctx, Wo, bo, lng, lnb, out);
}
